// SGNN_3573412790499
// MI455X (gfx1250) — compile-verified, matrix operand order checked by reading, not run on hardware
//
#include <hip/hip_runtime.h>

#define NNODE 64
#define HDIM  256
#define STEPS 2

typedef __attribute__((ext_vector_type(16))) __bf16 v16bf;
typedef __attribute__((ext_vector_type(8)))  float  v8f;

union BFU { __bf16 h; unsigned short u; };
__device__ inline unsigned short f2bf(float f) { BFU x; x.h = (__bf16)f; return x.u; }
__device__ inline float bf2f(unsigned short s) { BFU x; x.u = s; return (float)x.h; }
__device__ inline unsigned pk2(float a, float b) {
  return (unsigned)f2bf(a) | ((unsigned)f2bf(b) << 16);
}

union FragU { uint4 q[2]; v16bf v; };

// Load one 16x32 bf16 WMMA A/B fragment from a row-major bf16 matrix.
// Lane l: row = row_base + (l&15); K-halves: (l>>4)*8 and (l>>4)*8+16.
__device__ inline v16bf load_frag(const unsigned short* base, int row_base, int ld,
                                  int kofs, int lane) {
  int r  = row_base + (lane & 15);
  int kb = (lane >> 4) << 3;
  const unsigned short* p = base + r * ld + kofs + kb;
  FragU u;
  u.q[0] = *(const uint4*)(p);
  u.q[1] = *(const uint4*)(p + 16);
  return u.v;
}

__device__ inline v8f wmma_bf(v16bf a, v16bf b, v8f c) {
  return __builtin_amdgcn_wmma_f32_16x16x32_bf16(false, a, false, b, (short)0, c,
                                                 false, false);
}

__global__ void f2bf_kernel(const float* __restrict__ src,
                            unsigned short* __restrict__ dst, int n) {
  int i = blockIdx.x * blockDim.x + threadIdx.x;
  if (i < n) dst[i] = f2bf(src[i]);
}

// LDS layout (bytes):
//  sA    @ 0      : 64x128 bf16  (16384)
//  sH    @ 16384  : 64x256 bf16  (32768)
//  sIN   @ 49152  : 64x512 bf16  (65536)
//  sT    @ 114688 : HinT 256x64 + HoutT 256x64 bf16 (65536); aliased as hNew
//  star  @ 180224 : 256 f32
//  mask  @ 181248 : 64 f32
//  sim   @ 181504 : 64 f32
//  red   @ 181760 : f32
//  part  @ 181776 : 256 f32 partial sums
#define SMEM_BYTES 182816

__global__ __launch_bounds__(256)
void sgnn_fused(const float* __restrict__ A,
                const float* __restrict__ hidden_in,
                const float* __restrict__ gmask,
                const float* __restrict__ b_ih,  const float* __restrict__ b_hh,
                const float* __restrict__ b_iah, const float* __restrict__ b_oah,
                const float* __restrict__ b_in,  const float* __restrict__ b_out,
                const unsigned short* __restrict__ wih,
                const unsigned short* __restrict__ whh,
                const unsigned short* __restrict__ win,
                const unsigned short* __restrict__ wout,
                float* __restrict__ out_hidden,
                float* __restrict__ out_star) {
  extern __shared__ char smem[];
  unsigned short* sA  = (unsigned short*)(smem);
  unsigned short* sH  = (unsigned short*)(smem + 16384);
  unsigned short* sIN = (unsigned short*)(smem + 49152);
  unsigned short* sT  = (unsigned short*)(smem + 114688);
  unsigned short* sHn = sT;  // alias: HinT/HoutT dead once sIN is built
  float* sStar = (float*)(smem + 180224);
  float* sMask = (float*)(smem + 181248);
  float* sSim  = (float*)(smem + 181504);
  float* sRed  = (float*)(smem + 181760);
  float* sPart = (float*)(smem + 181776);

  const int b    = blockIdx.x;
  const int tid  = threadIdx.x;
  const int w    = tid >> 5;
  const int lane = tid & 31;
  const int ln   = lane & 15;
  const int lh   = lane >> 4;

  // warm L2 with the shared weights
  __builtin_prefetch(wih, 0, 1);
  __builtin_prefetch(whh, 0, 1);
  __builtin_prefetch(win, 0, 1);
  __builtin_prefetch(wout, 0, 1);

  // ---- load A, h, mask into LDS (f32 -> bf16) ----
  const float* Ab = A + (size_t)b * (NNODE * 2 * NNODE);
  for (int i = tid; i < NNODE * 2 * NNODE; i += 256) sA[i] = f2bf(Ab[i]);
  const float* Hb = hidden_in + (size_t)b * (NNODE * HDIM);
  for (int i = tid; i < NNODE * HDIM; i += 256) sH[i] = f2bf(Hb[i]);
  if (tid < NNODE) sMask[tid] = gmask[(size_t)b * NNODE + tid];
  __syncthreads();

  // ---- star = masked mean of hidden ----
  if (tid == 0) {
    float s = 0.f;
    for (int n = 0; n < NNODE; n++) s += sMask[n];
    sRed[0] = s;
  }
  __syncthreads();
  {
    float len = sRed[0];
    float s = 0.f;
    for (int n = 0; n < NNODE; n++) s += sMask[n] * bf2f(sH[n * HDIM + tid]);
    sStar[tid] = s / len;
  }
  __syncthreads();

  const float inv_sqrt_h = 0.0625f;  // 1/sqrt(256)

  for (int step = 0; step < STEPS; ++step) {
    // ---- P1: HinT/HoutT = (h @ W^T + bias)^T; one W fragment serves 4 m-tiles.
    //      Accumulators start at 0 (folds to inline SRC2=0); bias in epilogue. ----
    for (int t = w; t < 32; t += 8) {
      int sel = t >> 4;
      int nt  = t & 15;
      const unsigned short* W = sel ? wout : win;
      const float* bias = sel ? b_out : b_in;
      float bv = bias[nt * 16 + ln];
      v8f acc[4] = {};
#pragma unroll
      for (int kt = 0; kt < 8; ++kt) {
        v16bf bb = load_frag(W, nt * 16, HDIM, kt * 32, lane);
#pragma unroll
        for (int mt = 0; mt < 4; mt++) {
          v16bf a = load_frag(sH, mt * 16, HDIM, kt * 32, lane);
          acc[mt] = wmma_bf(a, bb, acc[mt]);
        }
      }
#pragma unroll
      for (int mt = 0; mt < 4; mt++) {
        uint4 pkd;
        pkd.x = pk2(acc[mt][0] + bv, acc[mt][1] + bv);
        pkd.y = pk2(acc[mt][2] + bv, acc[mt][3] + bv);
        pkd.z = pk2(acc[mt][4] + bv, acc[mt][5] + bv);
        pkd.w = pk2(acc[mt][6] + bv, acc[mt][7] + bv);
        unsigned short* dst = sT + sel * (HDIM * NNODE) + (nt * 16 + ln) * NNODE
                              + mt * 16 + lh * 8;
        *(uint4*)dst = pkd;
      }
    }
    __syncthreads();

    // ---- P2: inputs = [A_in@Hin + b_iah, A_out@Hout + b_oah]  (64x512 bf16) ----
    for (int t = w; t < 128; t += 8) {
      int sel = t >> 6;
      int tt = t & 63;
      int mt = tt & 3, nt = tt >> 2;
      const float* bias = sel ? b_oah : b_iah;
      float bv = bias[nt * 16 + ln];
      v8f acc = {};
      const unsigned short* Bt = sT + sel * (HDIM * NNODE);
#pragma unroll
      for (int kt = 0; kt < 2; ++kt) {
        v16bf a  = load_frag(sA, mt * 16, 2 * NNODE, sel * NNODE + kt * 32, lane);
        v16bf bb = load_frag(Bt, nt * 16, NNODE, kt * 32, lane);
        acc = wmma_bf(a, bb, acc);
      }
      int col  = sel * HDIM + nt * 16 + ln;
      int mrow = mt * 16 + lh * 8;
#pragma unroll
      for (int r = 0; r < 8; r++)
        sIN[(mrow + r) * (2 * HDIM) + col] = f2bf(acc[r] + bv);
    }
    __syncthreads();

    // ---- P3: gi = inputs @ w_ih^T, gh = h @ w_hh^T, fused GRU.
    //      Each wave owns 2 j-tiles; all 4 m-tiles batched (4x weight reuse).
    //      Zero-init accumulators; biases folded into the gate epilogue. ----
    for (int j = w; j < 16; j += 8) {
      v8f gi[4][3] = {}, gh[4][3] = {};
      float bRZ0 = b_ih[j * 16 + ln] + b_hh[j * 16 + ln];
      float bRZ1 = b_ih[HDIM + j * 16 + ln] + b_hh[HDIM + j * 16 + ln];
      float bN  = b_ih[2 * HDIM + j * 16 + ln];
      float cN  = b_hh[2 * HDIM + j * 16 + ln];
#pragma unroll
      for (int kt = 0; kt < 16; ++kt) {  // K = 512
        v16bf b0 = load_frag(wih, j * 16,        2 * HDIM, kt * 32, lane);
        v16bf b1 = load_frag(wih, (16 + j) * 16, 2 * HDIM, kt * 32, lane);
        v16bf b2 = load_frag(wih, (32 + j) * 16, 2 * HDIM, kt * 32, lane);
#pragma unroll
        for (int m = 0; m < 4; m++) {
          v16bf a = load_frag(sIN, m * 16, 2 * HDIM, kt * 32, lane);
          gi[m][0] = wmma_bf(a, b0, gi[m][0]);
          gi[m][1] = wmma_bf(a, b1, gi[m][1]);
          gi[m][2] = wmma_bf(a, b2, gi[m][2]);
        }
      }
#pragma unroll
      for (int kt = 0; kt < 8; ++kt) {  // K = 256
        v16bf b0 = load_frag(whh, j * 16,        HDIM, kt * 32, lane);
        v16bf b1 = load_frag(whh, (16 + j) * 16, HDIM, kt * 32, lane);
        v16bf b2 = load_frag(whh, (32 + j) * 16, HDIM, kt * 32, lane);
#pragma unroll
        for (int m = 0; m < 4; m++) {
          v16bf a = load_frag(sH, m * 16, HDIM, kt * 32, lane);
          gh[m][0] = wmma_bf(a, b0, gh[m][0]);
          gh[m][1] = wmma_bf(a, b1, gh[m][1]);
          gh[m][2] = wmma_bf(a, b2, gh[m][2]);
        }
      }
      int col = j * 16 + ln;
#pragma unroll
      for (int m = 0; m < 4; m++) {
        int mrow = m * 16 + lh * 8;
#pragma unroll
        for (int r = 0; r < 8; r++) {
          float rg = 1.f / (1.f + __expf(-(gi[m][0][r] + gh[m][0][r] + bRZ0)));
          float zg = 1.f / (1.f + __expf(-(gi[m][1][r] + gh[m][1][r] + bRZ1)));
          float ng = tanhf(gi[m][2][r] + bN + rg * (gh[m][2][r] + cN));
          float hv = bf2f(sH[(mrow + r) * HDIM + col]);
          sHn[(mrow + r) * HDIM + col] = f2bf(ng + zg * (hv - ng));
        }
      }
    }
    __syncthreads();
    for (int i = tid; i < NNODE * HDIM; i += 256) sH[i] = sHn[i];
    __syncthreads();

    // ---- P4: star attention (parallel quarter-dots + LDS reduction) ----
    {
      int n = tid & 63, q = tid >> 6;
      float s = 0.f;
      for (int k = q * 64; k < q * 64 + 64; k++)
        s += bf2f(sH[n * HDIM + k]) * sStar[k];
      sPart[tid] = s;
    }
    __syncthreads();
    if (tid < NNODE) {
      float s = sPart[tid] + sPart[tid + 64] + sPart[tid + 128] + sPart[tid + 192];
      sSim[tid] = 1.f / (1.f + __expf(-s * inv_sqrt_h));  // alpha
    }
    __syncthreads();
    for (int i = tid; i < NNODE * HDIM; i += 256) {
      int n = i >> 8, k = i & 255;
      float al = sSim[n];
      float hv = bf2f(sH[i]);
      sH[i] = f2bf((1.f - al) * hv + al * sStar[k]);
    }
    __syncthreads();
    {
      int n = tid & 63, q = tid >> 6;
      float s = 0.f;
      for (int k = q * 64; k < q * 64 + 64; k++)
        s += bf2f(sH[n * HDIM + k]) * sStar[k];
      sPart[tid] = s;
    }
    __syncthreads();
    if (tid < NNODE) {
      float s = sPart[tid] + sPart[tid + 64] + sPart[tid + 128] + sPart[tid + 192];
      sSim[tid] = __expf(s) * sMask[tid];
    }
    __syncthreads();
    if (tid == 0) {
      float s = 0.f;
      for (int n = 0; n < NNODE; n++) s += sSim[n];
      sRed[0] = s + 1e-24f;
    }
    __syncthreads();
    {
      float inv = 1.f / sRed[0];
      float s = 0.f;
      for (int n = 0; n < NNODE; n++) s += sSim[n] * bf2f(sH[n * HDIM + tid]);
      sStar[tid] = s * inv;
    }
    __syncthreads();
  }

  // ---- write outputs: hidden (f32) then star (f32) ----
  float* oh = out_hidden + (size_t)b * (NNODE * HDIM);
  for (int i = tid; i < NNODE * HDIM; i += 256) oh[i] = bf2f(sH[i]);
  out_star[(size_t)b * HDIM + tid] = sStar[tid];
}

extern "C" void kernel_launch(void* const* d_in, const int* in_sizes, int n_in,
                              void* d_out, int out_size, void* d_ws, size_t ws_size,
                              hipStream_t stream) {
  (void)in_sizes; (void)n_in; (void)out_size; (void)ws_size;
  const float* A      = (const float*)d_in[0];
  const float* hidden = (const float*)d_in[1];
  const float* gmask  = (const float*)d_in[2];
  const float* w_ih   = (const float*)d_in[3];
  const float* w_hh   = (const float*)d_in[4];
  const float* b_ih   = (const float*)d_in[5];
  const float* b_hh   = (const float*)d_in[6];
  const float* b_iah  = (const float*)d_in[7];
  const float* b_oah  = (const float*)d_in[8];
  const float* W_in   = (const float*)d_in[9];
  const float* b_in   = (const float*)d_in[10];
  const float* W_out  = (const float*)d_in[11];
  const float* b_out  = (const float*)d_in[12];

  unsigned short* ws     = (unsigned short*)d_ws;
  unsigned short* wih_b  = ws;             // 768*512
  unsigned short* whh_b  = ws + 393216;    // 768*256
  unsigned short* win_b  = ws + 589824;    // 256*256
  unsigned short* wout_b = ws + 655360;    // 256*256

  f2bf_kernel<<<(393216 + 255) / 256, 256, 0, stream>>>(w_ih, wih_b, 393216);
  f2bf_kernel<<<(196608 + 255) / 256, 256, 0, stream>>>(w_hh, whh_b, 196608);
  f2bf_kernel<<<(65536 + 255) / 256, 256, 0, stream>>>(W_in, win_b, 65536);
  f2bf_kernel<<<(65536 + 255) / 256, 256, 0, stream>>>(W_out, wout_b, 65536);

  float* out_hidden = (float*)d_out;
  float* out_star   = out_hidden + (size_t)2048 * NNODE * HDIM;

  sgnn_fused<<<2048, 256, SMEM_BYTES, stream>>>(
      A, hidden, gmask, b_ih, b_hh, b_iah, b_oah, b_in, b_out,
      wih_b, whh_b, win_b, wout_b, out_hidden, out_star);
}
